// GraphNN_10642928959898
// MI455X (gfx1250) — matrix operand order repaired and checked by reading, not compiled, not run on hardware
//
#include <hip/hip_runtime.h>
#include <math.h>

typedef __attribute__((ext_vector_type(16))) _Float16 v16h;
typedef __attribute__((ext_vector_type(8)))  _Float16 v8h;
typedef __attribute__((ext_vector_type(8)))  float    v8f;

#define TPB 256

// ---------------- utility kernels ----------------
__global__ void zero_f32(float* p, long n) {
  long i = (long)blockIdx.x * blockDim.x + threadIdx.x;
  if (i < n) p[i] = 0.f;
}
__global__ void zero_i32(int* p, long n) {
  long i = (long)blockIdx.x * blockDim.x + threadIdx.x;
  if (i < n) p[i] = 0;
}
__global__ void fill_f32(float* p, long n, float v) {
  long i = (long)blockIdx.x * blockDim.x + threadIdx.x;
  if (i < n) p[i] = v;
}
__global__ void exscan_serial(const int* in, int* out, int n) {
  if (blockIdx.x == 0 && threadIdx.x == 0) {
    int acc = 0;
    for (int i = 0; i < n; ++i) { out[i] = acc; acc += in[i]; }
    out[n] = acc;
  }
}

__device__ inline void atomicMaxF(float* addr, float val) {
  unsigned* a = (unsigned*)addr;
  unsigned old = *a;
  while (true) {
    float f = __uint_as_float(old);
    if (f >= val) break;
    unsigned assumed = old;
    old = atomicCAS(a, assumed, __float_as_uint(val));
    if (old == assumed) break;
  }
}

// ---------------- GCN: message scatter ----------------
// agg[dst][f] += x[src][f] * w[e]   over E directed edges, k features
__global__ void gcn_scatter(const float* __restrict__ x, const int* __restrict__ src,
                            const int* __restrict__ dst, const float* __restrict__ eattr,
                            float* __restrict__ agg, long E, int k) {
  long t = (long)blockIdx.x * blockDim.x + threadIdx.x;
  if (t >= E * (long)k) return;
  long e = t / k; int f = (int)(t % k);
  float w = eattr[e * 2];
  atomicAdd(&agg[(long)dst[e] * k + f], x[(long)src[e] * k + f] * w);
}

// pack [agg | x] (f32, n valid rows) -> f16 A matrix [npad][Kp], zero padded
__global__ void pack_gcn_A(const float* __restrict__ agg, const float* __restrict__ x,
                           _Float16* __restrict__ Af, int n, int npad, int k, int Kp) {
  long t = (long)blockIdx.x * blockDim.x + threadIdx.x;
  if (t >= (long)npad * Kp) return;
  int row = (int)(t / Kp), j = (int)(t % Kp);
  float v = 0.f;
  if (row < n) {
    if (j < k)          v = agg[(long)row * k + j];
    else if (j < 2 * k) v = x[(long)row * k + (j - k)];
  }
  Af[t] = (_Float16)v;
}

// pack weights (optionally two stacked matrices [w ; w2]) into WMMA B-fragment order:
//   Wp[(((ct*(Kp/32) + kb)*32) + lane)*16 + e] = Wrow[kb*32 + 16*(lane>>4) + e][ct*16 + (lane&15)]
// so each lane's 16 B-fragment halves are contiguous (two 16B vector loads in the GEMM).
__global__ void pack_W_frag(const float* __restrict__ w, const float* __restrict__ w2,
                            _Float16* __restrict__ Wp, int k, int o, int Kp) {
  long t = (long)blockIdx.x * blockDim.x + threadIdx.x;
  if (t >= (long)Kp * o) return;
  int kt = Kp >> 5;
  int e    = (int)(t & 15);
  int lane = (int)((t >> 4) & 31);
  long rest = t >> 9;
  int kb = (int)(rest % kt);
  int ct = (int)(rest / kt);
  int row = kb * 32 + 16 * (lane >> 4) + e;
  int col = ct * 16 + (lane & 15);
  float v = 0.f;
  if (row < k)                 v = w[(long)row * o + col];
  else if (w2 && row < 2 * k)  v = w2[(long)(row - k) * o + col];
  Wp[t] = (_Float16)v;
}

// ---------------- WMMA GEMM: out = act(A@W + b) ----------------
// A:  f16 [Mrows][K] row-major, Mrows%64==0, K%32==0 (zero-padded)
// Wp: f16 in fragment order from pack_W_frag. One wave per 64x16 output strip:
// 4 row tiles share each B fragment (4 independent WMMAs per K-step).
__global__ __launch_bounds__(32) void gemm_wmma(
    const _Float16* __restrict__ A, const _Float16* __restrict__ Wp,
    const float* __restrict__ bias, float* __restrict__ out,
    long Mrows, int K, int O, int act)
{
  int lane = threadIdx.x;
  long row0 = (long)blockIdx.x * 64;
  int nc = lane & 15;
  int h  = lane >> 4;
  int col = (blockIdx.y << 4) + nc;
  int kt = K >> 5;
  const _Float16* wp = Wp + ((long)blockIdx.y * kt * 32 + lane) * 16;
  const _Float16* ar = A + (row0 + nc) * (long)K;
  const long tstride = 16 * (long)K;
  v8f acc0 = {}, acc1 = {}, acc2 = {}, acc3 = {};
  for (int kb = 0; kb < kt; ++kb) {
    v8h b0 = *(const v8h*)(wp);
    v8h b1 = *(const v8h*)(wp + 8);
    wp += 512;
    v16h bv;
#pragma unroll
    for (int e = 0; e < 8; ++e) { bv[e] = b0[e]; bv[e + 8] = b1[e]; }
    const _Float16* a = ar + kb * 32;
    __builtin_prefetch(a + 32, 0, 1);
#pragma unroll
    for (int t = 0; t < 4; ++t) {
      v8h a0 = *(const v8h*)(a + 16 * h);
      v8h a1 = *(const v8h*)(a + 16 * h + 8);
      a += tstride;
      v16h av;
#pragma unroll
      for (int e = 0; e < 8; ++e) { av[e] = a0[e]; av[e + 8] = a1[e]; }
      v8f& acc = (t == 0 ? acc0 : (t == 1 ? acc1 : (t == 2 ? acc2 : acc3)));
      acc = __builtin_amdgcn_wmma_f32_16x16x32_f16(
          /*neg_a=*/false, av, /*neg_b=*/false, bv,
          /*c_mod=*/(short)0, acc, /*reuse_a=*/false, /*reuse_b=*/false);
    }
  }
  float bcol = bias ? bias[col] : 0.f;
#pragma unroll
  for (int t = 0; t < 4; ++t) {
    v8f acc = (t == 0 ? acc0 : (t == 1 ? acc1 : (t == 2 ? acc2 : acc3)));
#pragma unroll
    for (int v = 0; v < 8; ++v) {
      long r = row0 + t * 16 + 8 * h + v;  // D layout: VGPR v -> row v + 8*(lane>=16)
      float val = acc[v] + bcol;
      if (act) val = tanhf(val);
      out[r * O + col] = val;
    }
  }
}

// ---------------- stable compaction of kept edges ----------------
__global__ void keep_count(const int* __restrict__ src, const int* __restrict__ dst,
                           const int* __restrict__ det, int* __restrict__ blockCnt, long E) {
  __shared__ int cnt;
  if (threadIdx.x == 0) cnt = 0;
  __syncthreads();
  long i = (long)blockIdx.x * blockDim.x + threadIdx.x;
  int k = 0;
  if (i < E) { int s = src[i], d = dst[i]; k = (s > d) && det[s] && det[d]; }
  if (k) atomicAdd(&cnt, 1);
  __syncthreads();
  if (threadIdx.x == 0) blockCnt[blockIdx.x] = cnt;
}
__global__ void keep_scatter(const int* __restrict__ src, const int* __restrict__ dst,
                             const int* __restrict__ det, const int* __restrict__ blockOff,
                             int* __restrict__ idx, long E) {
  __shared__ int sh[TPB];
  int tid = threadIdx.x;
  long i = (long)blockIdx.x * blockDim.x + tid;
  int k = 0;
  if (i < E) { int s = src[i], d = dst[i]; k = (s > d) && det[s] && det[d]; }
  sh[tid] = k;
  __syncthreads();
  for (int off = 1; off < TPB; off <<= 1) {
    int v = (tid >= off) ? sh[tid - off] : 0;
    __syncthreads();
    sh[tid] += v;
    __syncthreads();
  }
  if (k) idx[blockOff[blockIdx.x] + sh[tid] - 1] = (int)i;
}

// ---------------- counting sort by e0 then per-bin sort by (e1, j) ----------------
__global__ void e0_hist(const int* __restrict__ idx, const int* __restrict__ src,
                        int* __restrict__ hist, int M) {
  int j = blockIdx.x * blockDim.x + threadIdx.x;
  if (j < M) atomicAdd(&hist[src[idx[j]]], 1);
}
__global__ void bin_scatter(const int* __restrict__ idx, const int* __restrict__ src,
                            const int* __restrict__ dst, const int* __restrict__ binStart,
                            int* __restrict__ binCnt, unsigned long long* __restrict__ keys, int M) {
  int j = blockIdx.x * blockDim.x + threadIdx.x;
  if (j >= M) return;
  int e = idx[j];
  int u = src[e], v = dst[e];
  int pos = binStart[u] + atomicAdd(&binCnt[u], 1);
  keys[pos] = (((unsigned long long)(unsigned)v) << 32) | (unsigned)j;
}
__global__ void bin_sort(const int* __restrict__ binStart, unsigned long long* __restrict__ keys, int n) {
  int u = blockIdx.x * blockDim.x + threadIdx.x;
  if (u >= n) return;
  int a = binStart[u], b = binStart[u + 1];
  for (int i = a + 1; i < b; ++i) {
    unsigned long long key = keys[i];
    int j = i - 1;
    while (j >= a && keys[j] > key) { keys[j + 1] = keys[j]; --j; }
    keys[j + 1] = key;
  }
}

// ---------------- edge feature construction (f16, K padded to 288) ----------------
__global__ void build_feat(const unsigned long long* __restrict__ keys, const int* __restrict__ idx,
                           const int* __restrict__ src, const int* __restrict__ dst,
                           const float* __restrict__ eattr, const int* __restrict__ batch,
                           const float* __restrict__ x3, _Float16* __restrict__ feat,
                           int* __restrict__ es0, int* __restrict__ es1, int* __restrict__ ebArr,
                           float* __restrict__ clsArr, int Kp, int C) {
  long s = blockIdx.x;
  int j = (int)(keys[s] & 0xffffffffu);
  int e = idx[j];
  int u = src[e], v = dst[e];
  float w = eattr[(long)e * 2], cls = eattr[(long)e * 2 + 1];
  if (threadIdx.x == 0) { es0[s] = u; es1[s] = v; ebArr[s] = batch[u]; clsArr[s] = cls; }
  _Float16* row = feat + s * (long)Kp;
  for (int c = threadIdx.x; c < Kp; c += blockDim.x) {
    float val = 0.f;
    if (c < C)                 val = x3[(long)u * C + c];
    else if (c == C)           val = w;
    else if (c == C + 1)       val = (cls == 0.f) ? 1.f : 0.f;
    else if (c == C + 2)       val = (cls == 1.f) ? 1.f : 0.f;
    else if (c < 2 * C + 3)    val = x3[(long)v * C + (c - (C + 3))];
    row[c] = (_Float16)val;
  }
}

// ---------------- segment (batch) reductions over sorted eb ----------------
__global__ void seg_bounds(const int* __restrict__ eb, int* __restrict__ sI, int* __restrict__ eI, int M) {
  int s = blockIdx.x * blockDim.x + threadIdx.x;
  if (s >= M) return;
  int b = eb[s];
  if (s == 0 || eb[s - 1] != b) sI[b] = s;
  if (s == M - 1 || eb[s + 1] != b) eI[b] = s + 1;
}
__global__ void cnt_k(const int* __restrict__ sI, const int* __restrict__ eI, float* __restrict__ cntf, int B) {
  int b = threadIdx.x;
  if (b >= B) return;
  int c = eI[b] - sI[b];
  cntf[b] = (float)(c > 0 ? c : 1);
}
// run-length accumulation: one atomic per (segment boundary, column)
__global__ void segsum_sorted(const float* __restrict__ h, const int* __restrict__ eb,
                              float* __restrict__ sums, long M, int C, int R) {
  int c = threadIdx.x;
  long r0 = (long)blockIdx.x * R;
  long r1 = r0 + R; if (r1 > M) r1 = M;
  if (r0 >= r1) return;
  int cur = eb[r0];
  float acc = 0.f;
  for (long r = r0; r < r1; ++r) {
    int b = eb[r];
    if (b != cur) { atomicAdd(&sums[(long)cur * C + c], acc); acc = 0.f; cur = b; }
    acc += h[r * C + c];
  }
  atomicAdd(&sums[(long)cur * C + c], acc);
}
__global__ void mean_k(const float* __restrict__ sums, const float* __restrict__ cntf,
                       float* __restrict__ meanb, int C) {
  int t = blockIdx.x * blockDim.x + threadIdx.x;
  if (t >= 64 * C) return;
  meanb[t] = sums[t] / cntf[t / C];
}
// h0 = h - na*mean[eb]; h := h0 (in place); varsums += h0*h0 per (eb,c)
__global__ void h0_varsum(float* __restrict__ h, const int* __restrict__ eb,
                          const float* __restrict__ meanb, const float* __restrict__ na,
                          float* __restrict__ vars, long M, int C, int R) {
  int c = threadIdx.x;
  long r0 = (long)blockIdx.x * R;
  long r1 = r0 + R; if (r1 > M) r1 = M;
  if (r0 >= r1) return;
  int cur = eb[r0];
  float acc = 0.f;
  float nac = na[c];
  for (long r = r0; r < r1; ++r) {
    int b = eb[r];
    if (b != cur) { atomicAdd(&vars[(long)cur * C + c], acc); acc = 0.f; cur = b; }
    float val = h[r * C + c] - nac * meanb[(long)b * C + c];
    h[r * C + c] = val;
    acc += val * val;
  }
  atomicAdd(&vars[(long)cur * C + c], acc);
}
__global__ void apply_norm(const float* __restrict__ h0, const int* __restrict__ eb,
                           const float* __restrict__ vars, const float* __restrict__ cntf,
                           const float* __restrict__ ng, const float* __restrict__ nb,
                           _Float16* __restrict__ outf, long M, int C) {
  long t = (long)blockIdx.x * blockDim.x + threadIdx.x;
  if (t >= M * (long)C) return;
  long r = t / C; int c = (int)(t % C);
  int b = eb[r];
  float v = vars[(long)b * C + c] / cntf[b];
  outf[t] = (_Float16)(ng[c] * h0[t] / sqrtf(v + 1e-5f) + nb[c]);
}

// ---------------- output head ----------------
__global__ void logit_k(const _Float16* __restrict__ f, const float* __restrict__ ow,
                        const float* __restrict__ ob, float* __restrict__ logit, long M, int C) {
  long s = (long)blockIdx.x * blockDim.x + threadIdx.x;
  if (s >= M) return;
  float acc = ob[0];
  const _Float16* row = f + s * (long)C;
  for (int c = 0; c < C; ++c) acc += (float)row[c] * ow[c];
  logit[s] = acc;
}
__global__ void pair_k(const float* __restrict__ logit, const int* __restrict__ es0,
                       const int* __restrict__ es1, const float* __restrict__ clsArr,
                       const int* __restrict__ ebArr, float* __restrict__ out,
                       float* __restrict__ minv, int* __restrict__ eb2, float* __restrict__ mmax, int Pp) {
  int p = blockIdx.x * blockDim.x + threadIdx.x;
  if (p >= Pp) return;
  float l0 = logit[2 * p], l1 = logit[2 * p + 1];
  int mi = (l1 < l0) ? 1 : 0;           // argmin, first index on tie
  float mv = mi ? l1 : l0;
  minv[p] = mv;
  int b = ebArr[2 * p];
  eb2[p] = b;
  out[p]            = (float)es0[2 * p];        // edges_out row 0
  out[Pp + p]       = (float)es1[2 * p];        // edges_out row 1
  out[3 * Pp + p]   = clsArr[2 * p + mi];       // edge_classes
  atomicMaxF(&mmax[b], mv);
}
__global__ void exp_k(const float* __restrict__ minv, const int* __restrict__ eb2,
                      const float* __restrict__ mmax, float* __restrict__ exv,
                      float* __restrict__ ssum, int Pp) {
  int p = blockIdx.x * blockDim.x + threadIdx.x;
  if (p >= Pp) return;
  float e = expf(minv[p] - mmax[eb2[p]]);
  exv[p] = e;
  atomicAdd(&ssum[eb2[p]], e);
}
__global__ void soft_k(const float* __restrict__ exv, const int* __restrict__ eb2,
                       const float* __restrict__ ssum, float* __restrict__ out, int Pp) {
  int p = blockIdx.x * blockDim.x + threadIdx.x;
  if (p >= Pp) return;
  out[2 * Pp + p] = exv[p] / (ssum[eb2[p]] + 1e-16f);
}

static inline dim3 g1(long n) { return dim3((unsigned)((n + TPB - 1) / TPB)); }

extern "C" void kernel_launch(void* const* d_in, const int* in_sizes, int n_in,
                              void* d_out, int out_size, void* d_ws, size_t ws_size,
                              hipStream_t stream) {
  (void)n_in; (void)out_size; (void)ws_size;
  const float* x_in  = (const float*)d_in[0];
  const int*   edges = (const int*)d_in[1];
  const float* eattr = (const float*)d_in[2];
  const int*   det   = (const int*)d_in[3];
  const int*   batch = (const int*)d_in[4];
  const float* gwr[3] = {(const float*)d_in[5],  (const float*)d_in[8],  (const float*)d_in[11]};
  const float* gws[3] = {(const float*)d_in[6],  (const float*)d_in[9],  (const float*)d_in[12]};
  const float* gb[3]  = {(const float*)d_in[7],  (const float*)d_in[10], (const float*)d_in[13]};
  const float* d0w = (const float*)d_in[14]; const float* d0b = (const float*)d_in[15];
  const float* n0g = (const float*)d_in[16]; const float* n0b = (const float*)d_in[17];
  const float* n0a = (const float*)d_in[18];
  const float* d1w = (const float*)d_in[19]; const float* d1b = (const float*)d_in[20];
  const float* n1g = (const float*)d_in[21]; const float* n1b = (const float*)d_in[22];
  const float* n1a = (const float*)d_in[23];
  const float* ow  = (const float*)d_in[24]; const float* ob = (const float*)d_in[25];
  float* out = (float*)d_out;

  const int  N    = in_sizes[0] / 5;       // 20000
  const int  Npad = ((N + 63) / 64) * 64;  // 20032 (64-row GEMM strips)
  const long E    = in_sizes[1] / 2;       // 4P = 1.2M directed edges
  const long M    = E / 2;                 // 2P kept+sorted edges (64-divisible)
  const int  Pp   = (int)(M / 2);          // P output pairs
  const long nb   = (E + TPB - 1) / TPB;

  const int* src = edges;
  const int* dst = edges + E;

  // ---- workspace carve (256B aligned) ----
  char* wsB = (char*)d_ws;
  size_t off = 0;
  auto carve = [&](size_t bytes) -> char* {
    char* p = wsB + off;
    off += (bytes + 255) & ~(size_t)255;
    return p;
  };
  float*    agg    = (float*)carve((size_t)N * 128 * 4);
  float*    xb1    = (float*)carve((size_t)Npad * 128 * 4);
  float*    xb2    = (float*)carve((size_t)Npad * 128 * 4);
  _Float16* Agcn   = (_Float16*)carve((size_t)Npad * 128 * 2);
  _Float16* Wf     = (_Float16*)carve((size_t)288 * 128 * 2);
  int*      blockCnt = (int*)carve((size_t)(nb + 1) * 4);
  int*      blockOff = (int*)carve((size_t)(nb + 1) * 4);
  int*      idx    = (int*)carve((size_t)M * 4);
  int*      hist   = (int*)carve((size_t)(N + 1) * 4);
  int*      binStart = (int*)carve((size_t)(N + 1) * 4);
  int*      binCnt = (int*)carve((size_t)N * 4);
  unsigned long long* keys = (unsigned long long*)carve((size_t)M * 8);
  _Float16* feat16 = (_Float16*)carve((size_t)M * 288 * 2);
  float*    hbuf   = (float*)carve((size_t)M * 128 * 4);
  int*      es0    = (int*)carve((size_t)M * 4);
  int*      es1    = (int*)carve((size_t)M * 4);
  int*      ebArr  = (int*)carve((size_t)M * 4);
  float*    clsArr = (float*)carve((size_t)M * 4);
  float*    logit  = (float*)carve((size_t)M * 4);
  float*    minv   = (float*)carve((size_t)Pp * 4);
  float*    exv    = (float*)carve((size_t)Pp * 4);
  int*      eb2    = (int*)carve((size_t)Pp * 4);
  float*    mmax   = (float*)carve(64 * 4);
  float*    ssum   = (float*)carve(64 * 4);
  float*    sums   = (float*)carve(64 * 128 * 4);
  float*    meanb  = (float*)carve(64 * 128 * 4);
  float*    vars   = (float*)carve(64 * 128 * 4);
  float*    cntf   = (float*)carve(64 * 4);
  int*      startI = (int*)carve(64 * 4);
  int*      endI   = (int*)carve(64 * 4);
  // aliases (lifetimes do not overlap)
  _Float16* featn1 = feat16;                       // after GEMM1 consumed feat16
  _Float16* featn2 = feat16 + (size_t)M * 128;     // disjoint from featn1
  float*    h2     = hbuf;                          // after layer-0 norm consumed h0

  // ================= GCN layers =================
  int dims[4] = {5, 32, 64, 128};
  const float* xcur = x_in;
  float* xping = xb1;
  float* xpong = xb2;
  for (int l = 0; l < 3; ++l) {
    int k = dims[l], o = dims[l + 1];
    int Kp = ((2 * k + 31) / 32) * 32;  // 32, 64, 128
    zero_f32<<<g1((long)N * k), TPB, 0, stream>>>(agg, (long)N * k);
    gcn_scatter<<<g1(E * (long)k), TPB, 0, stream>>>(xcur, src, dst, eattr, agg, E, k);
    pack_gcn_A<<<g1((long)Npad * Kp), TPB, 0, stream>>>(agg, xcur, Agcn, N, Npad, k, Kp);
    pack_W_frag<<<g1((long)Kp * o), TPB, 0, stream>>>(gwr[l], gws[l], Wf, k, o, Kp);
    gemm_wmma<<<dim3((unsigned)(Npad / 64), (unsigned)(o / 16)), 32, 0, stream>>>(
        Agcn, Wf, gb[l], xping, (long)Npad, Kp, o, 1);
    xcur = xping;
    float* t = xping; xping = xpong; xpong = t;
  }
  const float* x3 = xcur;  // [Npad][128], rows >= N are dead

  // ================= edge compaction (stable) =================
  keep_count<<<dim3((unsigned)nb), TPB, 0, stream>>>(src, dst, det, blockCnt, E);
  exscan_serial<<<1, 1, 0, stream>>>(blockCnt, blockOff, (int)nb);
  keep_scatter<<<dim3((unsigned)nb), TPB, 0, stream>>>(src, dst, det, blockOff, idx, E);

  // ================= counting sort by (e0, e1, stable j) =================
  zero_i32<<<g1(N + 1), TPB, 0, stream>>>(hist, N + 1);
  e0_hist<<<g1(M), TPB, 0, stream>>>(idx, src, hist, (int)M);
  exscan_serial<<<1, 1, 0, stream>>>(hist, binStart, N);
  zero_i32<<<g1(N), TPB, 0, stream>>>(binCnt, N);
  bin_scatter<<<g1(M), TPB, 0, stream>>>(idx, src, dst, binStart, binCnt, keys, (int)M);
  bin_sort<<<g1(N), TPB, 0, stream>>>(binStart, keys, N);

  // ================= features + MLP =================
  build_feat<<<dim3((unsigned)M), 128, 0, stream>>>(keys, idx, src, dst, eattr, batch, x3,
                                                    feat16, es0, es1, ebArr, clsArr, 288, 128);
  // batch counts (eb sorted; same for both norm layers)
  zero_i32<<<1, 64, 0, stream>>>(startI, 64);
  zero_i32<<<1, 64, 0, stream>>>(endI, 64);
  seg_bounds<<<g1(M), TPB, 0, stream>>>(ebArr, startI, endI, (int)M);
  cnt_k<<<1, 64, 0, stream>>>(startI, endI, cntf, 64);

  // --- dense layer 0: h = tanh(feat @ d0_w + d0_b), K=259 padded to 288 ---
  pack_W_frag<<<g1((long)288 * 128), TPB, 0, stream>>>(d0w, (const float*)0, Wf, 259, 128, 288);
  gemm_wmma<<<dim3((unsigned)(M / 64), 8), 32, 0, stream>>>(feat16, Wf, d0b, hbuf, M, 288, 128, 1);
  // --- norm layer 0 (C=128) ---
  {
    int C = 128, R = 512;
    zero_f32<<<g1(64 * C), TPB, 0, stream>>>(sums, 64 * C);
    segsum_sorted<<<dim3((unsigned)((M + R - 1) / R)), C, 0, stream>>>(hbuf, ebArr, sums, M, C, R);
    mean_k<<<g1(64 * C), TPB, 0, stream>>>(sums, cntf, meanb, C);
    zero_f32<<<g1(64 * C), TPB, 0, stream>>>(vars, 64 * C);
    h0_varsum<<<dim3((unsigned)((M + R - 1) / R)), C, 0, stream>>>(hbuf, ebArr, meanb, n0a, vars, M, C, R);
    apply_norm<<<g1(M * (long)C), TPB, 0, stream>>>(hbuf, ebArr, vars, cntf, n0g, n0b, featn1, M, C);
  }
  // --- dense layer 1: h2 = tanh(featn1 @ d1_w + d1_b), K=128, O=64 ---
  pack_W_frag<<<g1((long)128 * 64), TPB, 0, stream>>>(d1w, (const float*)0, Wf, 128, 64, 128);
  gemm_wmma<<<dim3((unsigned)(M / 64), 4), 32, 0, stream>>>(featn1, Wf, d1b, h2, M, 128, 64, 1);
  // --- norm layer 1 (C=64) ---
  {
    int C = 64, R = 512;
    zero_f32<<<g1(64 * C), TPB, 0, stream>>>(sums, 64 * C);
    segsum_sorted<<<dim3((unsigned)((M + R - 1) / R)), C, 0, stream>>>(h2, ebArr, sums, M, C, R);
    mean_k<<<g1(64 * C), TPB, 0, stream>>>(sums, cntf, meanb, C);
    zero_f32<<<g1(64 * C), TPB, 0, stream>>>(vars, 64 * C);
    h0_varsum<<<dim3((unsigned)((M + R - 1) / R)), C, 0, stream>>>(h2, ebArr, meanb, n1a, vars, M, C, R);
    apply_norm<<<g1(M * (long)C), TPB, 0, stream>>>(h2, ebArr, vars, cntf, n1g, n1b, featn2, M, C);
  }

  // ================= head: logit, pairwise min, per-batch softmax =================
  logit_k<<<g1(M), TPB, 0, stream>>>(featn2, ow, ob, logit, M, 64);
  fill_f32<<<1, 64, 0, stream>>>(mmax, 64, -3.4028235e38f);
  zero_f32<<<1, 64, 0, stream>>>(ssum, 64);
  pair_k<<<g1(Pp), TPB, 0, stream>>>(logit, es0, es1, clsArr, ebArr, out, minv, eb2, mmax, Pp);
  exp_k<<<g1(Pp), TPB, 0, stream>>>(minv, eb2, mmax, exv, ssum, Pp);
  soft_k<<<g1(Pp), TPB, 0, stream>>>(exv, eb2, ssum, out, Pp);
}
